// SetConvGridEncoder_21105469292680
// MI455X (gfx1250) — matrix operand order repaired and checked by reading, not compiled, not run on hardware
//
#include <hip/hip_runtime.h>

typedef __attribute__((ext_vector_type(16))) _Float16 v16h;
typedef __attribute__((ext_vector_type(8)))  float    v8f;

#define N_PTS  4096   // data points per batch (K dimension)
#define M_TOT  4096   // grid points (64*64)
#define DZ     16
#define BATCH  8

// workspace layout
#define WS_HDR_OFF   0                        // 2 floats: inv_ls0, inv_ls1
#define WS_XS_OFF    1024                     // [8,4096,2] f32, pre-scaled x
#define WS_XS_BYTES  (BATCH * N_PTS * 2 * 4)  // 256 KB
#define WS_ZT_OFF    (WS_XS_OFF + WS_XS_BYTES)          // 32B-aligned
#define WS_ZT_BYTES  (BATCH * DZ * N_PTS * 2)           // 1 MB, f16 z^T
#define WS_NEEDED    (WS_ZT_OFF + WS_ZT_BYTES)

// ---------------- pre-passes ----------------

// inv_ls = 1 / (1e-5 + softplus(p))
__global__ void prep_hdr_kernel(const float* __restrict__ lsp, float* __restrict__ hdr) {
    if (threadIdx.x < 2)
        hdr[threadIdx.x] = 1.0f / (1e-5f + log1pf(__expf(lsp[threadIdx.x])));
}

// xs = x * inv_ls (per dim)
__global__ void prep_xs_kernel(const float* __restrict__ x, const float* __restrict__ lsp,
                               float* __restrict__ xs, int total) {
    const int i = blockIdx.x * blockDim.x + threadIdx.x;
    if (i < total) {
        const float inv = 1.0f / (1e-5f + log1pf(__expf(lsp[i & 1])));
        xs[i] = x[i] * inv;
    }
}

// zt[b][n][k] = (f16) z[b][k][n]  -- B-fragment-friendly transpose
__global__ __launch_bounds__(256) void prep_zt_kernel(const float* __restrict__ z,
                                                      _Float16* __restrict__ zt) {
    const int t  = blockIdx.x * blockDim.x + threadIdx.x;   // 0 .. 32767
    const int b  = t >> 12;
    const int r  = t & 4095;
    const int n  = r >> 8;                                   // 0..15
    const int k0 = (r & 255) * 16;
    const float*    zb = z  + (size_t)b * N_PTS * DZ;
    _Float16*       o  = zt + ((size_t)b * DZ + n) * N_PTS + k0;
#pragma unroll
    for (int j = 0; j < 16; ++j)
        o[j] = (_Float16)zb[(size_t)(k0 + j) * DZ + n];
}

// ---------------- main fused kernel ----------------
// One wave32 computes a 16x16 tile: D[16 grid rows x 16 channels] =
// sum_K W[16x32, generated in-register] x Z[32x16].
__global__ __launch_bounds__(256) void setconv_wmma_kernel(
    const float* __restrict__ grid,      // [4096,2]
    const float* __restrict__ hdr,       // inv_ls0, inv_ls1
    const float* __restrict__ xs,        // [8,4096,2] pre-scaled
    const _Float16* __restrict__ zt,     // [8,16,4096] f16 transposed
    float* __restrict__ zout)            // [8,4096,16]
{
    const int lane   = threadIdx.x & 31;
    const int waveId = blockIdx.x * 8 + (threadIdx.x >> 5);  // 0..2047
    const int b      = waveId >> 8;
    const int mbase  = (waveId & 255) << 4;

    const float inv0 = hdr[0];
    const float inv1 = hdr[1];

    // grid point owned by this lane, pre-scaled by 1/ls
    const int   mrow = mbase + (lane & 15);
    const float G0   = grid[mrow * 2 + 0] * inv0;
    const float G1   = grid[mrow * 2 + 1] * inv1;

    const int hi   = (lane >> 4) & 1;
    const int ncol = lane & 15;
    const float*    xsb   = xs + (size_t)b * N_PTS * 2;
    const _Float16* ztrow = zt + ((size_t)b * DZ + ncol) * N_PTS;

    v8f acc = {};

#pragma unroll 2
    for (int k0 = 0; k0 < N_PTS; k0 += 32) {
        __builtin_prefetch(xsb + (size_t)(k0 + 64) * 2, 0, 0);  // global_prefetch_b8

        // A fragment: contiguous k order (k = 16*hi + s), same as B.
        // VGPR j(0..7): K = k0 + hi*16 + 2j,+1
        v16h a;
#pragma unroll
        for (int j = 0; j < 8; ++j) {
            const int    kk = k0 + (hi ? 16 : 0) + 2 * j;
            const float4 u  = *reinterpret_cast<const float4*>(xsb + (size_t)kk * 2);
            float d0 = G0 - u.x;
            float d1 = G1 - u.y;
            const float w0 = __expf(-0.5f * __builtin_fmaf(d0, d0, d1 * d1));
            d0 = G0 - u.z;
            d1 = G1 - u.w;
            const float w1 = __expf(-0.5f * __builtin_fmaf(d0, d0, d1 * d1));
            a[2 * j + 0] = (_Float16)w0;
            a[2 * j + 1] = (_Float16)w1;
        }

        // B fragment: 16 consecutive f16 K-values for this lane's column -> 2x b128
        const v16h bf = *reinterpret_cast<const v16h*>(ztrow + k0 + hi * 16);

        acc = __builtin_amdgcn_wmma_f32_16x16x32_f16(
            false, a, false, bf, (short)0, acc, false, false);
    }

    // D: VGPR r -> M = r + hi*8, N = ncol
    float* ob = zout + ((size_t)b * M_TOT + mbase) * DZ;
#pragma unroll
    for (int r = 0; r < 8; ++r)
        ob[(size_t)(r + hi * 8) * DZ + ncol] = acc[r];
}

// ---------------- fallback (no workspace): round-1 direct kernel ----------------
__global__ __launch_bounds__(256) void setconv_wmma_direct_kernel(
    const float* __restrict__ x, const float* __restrict__ z,
    const float* __restrict__ lsp, const float* __restrict__ grid,
    float* __restrict__ zout)
{
    const int lane   = threadIdx.x & 31;
    const int waveId = blockIdx.x * 8 + (threadIdx.x >> 5);
    const int b      = waveId >> 8;
    const int mbase  = (waveId & 255) << 4;

    const float inv0 = 1.0f / (1e-5f + log1pf(__expf(lsp[0])));
    const float inv1 = 1.0f / (1e-5f + log1pf(__expf(lsp[1])));

    const int   mrow = mbase + (lane & 15);
    const float gx = grid[mrow * 2 + 0], gy = grid[mrow * 2 + 1];
    const int hi = (lane >> 4) & 1, ncol = lane & 15;
    const float* xb = x + (size_t)b * N_PTS * 2;
    const float* zb = z + (size_t)b * N_PTS * DZ;

    v8f acc = {};
    for (int k0 = 0; k0 < N_PTS; k0 += 32) {
        v16h a;
#pragma unroll
        for (int j = 0; j < 8; ++j) {
            const int    kk = k0 + (hi ? 16 : 0) + 2 * j;
            const float4 xv = *reinterpret_cast<const float4*>(xb + (size_t)kk * 2);
            float d0 = (gx - xv.x) * inv0, d1 = (gy - xv.y) * inv1;
            const float w0 = __expf(-0.5f * __builtin_fmaf(d0, d0, d1 * d1));
            d0 = (gx - xv.z) * inv0; d1 = (gy - xv.w) * inv1;
            const float w1 = __expf(-0.5f * __builtin_fmaf(d0, d0, d1 * d1));
            a[2 * j + 0] = (_Float16)w0;
            a[2 * j + 1] = (_Float16)w1;
        }
        v16h bf;
        const int kb = k0 + (hi ? 16 : 0);
#pragma unroll
        for (int j = 0; j < 16; ++j)
            bf[j] = (_Float16)zb[(size_t)(kb + j) * DZ + ncol];
        acc = __builtin_amdgcn_wmma_f32_16x16x32_f16(
            false, a, false, bf, (short)0, acc, false, false);
    }
    float* ob = zout + ((size_t)b * M_TOT + mbase) * DZ;
#pragma unroll
    for (int r = 0; r < 8; ++r)
        ob[(size_t)(r + hi * 8) * DZ + ncol] = acc[r];
}

// x_grid = broadcast grid [64,64,2] -> [8,64,64,2]
__global__ void xgrid_broadcast_kernel(const float* __restrict__ grid,
                                       float* __restrict__ out, int total) {
    const int i = blockIdx.x * blockDim.x + threadIdx.x;
    if (i < total) out[i] = grid[i & (64 * 64 * 2 - 1)];
}

extern "C" void kernel_launch(void* const* d_in, const int* in_sizes, int n_in,
                              void* d_out, int out_size, void* d_ws, size_t ws_size,
                              hipStream_t stream) {
    const float* x    = (const float*)d_in[0];
    const float* z    = (const float*)d_in[1];
    const float* lsp  = (const float*)d_in[2];
    const float* grid = (const float*)d_in[3];

    float* out    = (float*)d_out;
    float* x_grid = out;
    float* z_grid = out + BATCH * M_TOT * 2;

    xgrid_broadcast_kernel<<<BATCH * M_TOT * 2 / 256, 256, 0, stream>>>(
        grid, x_grid, BATCH * M_TOT * 2);

    if (ws_size >= (size_t)WS_NEEDED) {
        char* ws  = (char*)d_ws;
        float*    hdr = (float*)(ws + WS_HDR_OFF);
        float*    xs  = (float*)(ws + WS_XS_OFF);
        _Float16* zt  = (_Float16*)(ws + WS_ZT_OFF);

        prep_hdr_kernel<<<1, 32, 0, stream>>>(lsp, hdr);
        prep_xs_kernel<<<BATCH * N_PTS * 2 / 256, 256, 0, stream>>>(
            x, lsp, xs, BATCH * N_PTS * 2);
        prep_zt_kernel<<<BATCH * DZ * (N_PTS / 16) / 256, 256, 0, stream>>>(z, zt);

        setconv_wmma_kernel<<<256, 256, 0, stream>>>(grid, hdr, xs, zt, z_grid);
    } else {
        setconv_wmma_direct_kernel<<<256, 256, 0, stream>>>(x, z, lsp, grid, z_grid);
    }
}
